// TGCNLayer_55972013801819
// MI455X (gfx1250) — matrix operand order repaired and checked by reading, not compiled, not run on hardware
//
#include <hip/hip_runtime.h>
#include <hip/hip_bf16.h>

#define NNODES 25000
#define KNB 16
#define FDIM 64
#define ADIM 64
#define DWDIM 64
#define INK 560
#define INKP 576
#define OUTF 64
#define JSTR 72   // padded bf16 row stride for 16x64 staging tiles (conflict-free)
#define RSTR 68   // padded bf16 row stride for atten2 row staging

typedef __attribute__((ext_vector_type(16))) __bf16 v16bf;
typedef __attribute__((ext_vector_type(8)))  float  v8f;

__device__ __forceinline__ __bf16 f2bf(float x) {
  union { float f; unsigned u; } a; a.f = x;
  unsigned r = (a.u + 0x7FFFu + ((a.u >> 16) & 1u)) >> 16;
  union { unsigned short s; __bf16 b; } o; o.s = (unsigned short)r;
  return o.b;
}
__device__ __forceinline__ float bf2f(__bf16 b) {
  union { unsigned short s; __bf16 b; } i; i.b = b;
  union { unsigned u; float f; } o; o.u = ((unsigned)i.s) << 16;
  return o.f;
}
__device__ __forceinline__ float reluf(float x) { return x > 0.f ? x : 0.f; }

__device__ __forceinline__ v8f wmma_bf16(v16bf a, v16bf b, v8f c) {
  return __builtin_amdgcn_wmma_f32_16x16x32_bf16(false, a, false, b, (short)0, c,
                                                 false, false);
}

// A-fragment (16x32 bf16) from row-major [16][lda] at K-offset k0.
__device__ __forceinline__ v16bf a_frag_lds(const __bf16* s, int lda, int k0, int lane) {
  const int half = lane >> 4;
  const int m = lane & 15;
  v16bf a;
#pragma unroll
  for (int h = 0; h < 16; ++h) {
    const int g = h >> 1, b = h & 1;
    int k = (g < 4) ? (2 * g + b) : (16 + 2 * (g - 4) + b);
    k += half ? 8 : 0;
    a[h] = s[m * lda + k0 + k];
  }
  return a;
}

// B-fragment (32x16 bf16) from row-major [K][lda] at (k0, n0).
__device__ __forceinline__ v16bf b_frag_lds(const __bf16* s, int lda, int k0, int n0,
                                            int lane) {
  const int n = n0 + (lane & 15);
  const int kh = k0 + ((lane >> 4) << 3);
  v16bf bf;
#pragma unroll
  for (int h = 0; h < 16; ++h) bf[h] = s[(kh + h + (h >= 8 ? 8 : 0)) * lda + n];
  return bf;
}

// ---------------------------------------------------------------------------
// atten1: per node, gather K=16 neighbor rows, attention via bf16 WMMA.
// ---------------------------------------------------------------------------
__global__ __launch_bounds__(256) void atten1_kernel(
    const float* __restrict__ ev, const float* __restrict__ ej,
    const float* __restrict__ ew, const int* __restrict__ vj,
    const int* __restrict__ vw, const float* __restrict__ W1,
    const float* __restrict__ W2, const float* __restrict__ bias,
    const float* __restrict__ vvec, float* __restrict__ outp) {
  __shared__ __bf16 sW1a[64 * 64];
  __shared__ __bf16 sW1b[64 * 64];
  __shared__ __bf16 sW2[64 * 64];
  __shared__ float sBias[64], sVvec[64];
  __shared__ __bf16 sJ[8][16 * JSTR];
  __shared__ __bf16 sWt[8][16 * JSTR];
  __shared__ float sAv[8][64];
  __shared__ float sX[8][16];

  const int tid = threadIdx.x;
  for (int i = tid; i < 64 * 64; i += 256) {
    sW1a[i] = f2bf(W1[i]);
    sW1b[i] = f2bf(W1[64 * 64 + i]);
    sW2[i] = f2bf(W2[i]);
  }
  if (tid < 64) { sBias[tid] = bias[tid]; sVvec[tid] = vvec[tid]; }
  __syncthreads();

  const int lane = tid & 31;
  const int w = tid >> 5;

  v16bf BW1[2][4], BW2[2][4];
#pragma unroll
  for (int kc = 0; kc < 2; ++kc)
#pragma unroll
    for (int nt = 0; nt < 4; ++nt) {
      BW1[kc][nt] = b_frag_lds(sW1b, 64, kc * 32, nt * 16, lane);
      BW2[kc][nt] = b_frag_lds(sW2, 64, kc * 32, nt * 16, lane);
    }

  const int gw = blockIdx.x * 8 + w;
  const int nw = gridDim.x * 8;
  for (int n = gw; n < NNODES; n += nw) {
    {
      const int k = lane >> 1;
      const int cb = (lane & 1) * 32;
      const int ji = vj[n * KNB + k];
      const int wi = vw[n * KNB + k];
      const float* jr = (ji == 0) ? nullptr : (ej + (size_t)(ji - 1) * FDIM);
      const float* wr = (wi == 0) ? nullptr : (ew + (size_t)(wi - 1) * DWDIM);
#pragma unroll
      for (int q4 = 0; q4 < 8; ++q4) {
        float4 aj = jr ? ((const float4*)jr)[cb / 4 + q4] : make_float4(0.f, 0.f, 0.f, 0.f);
        float4 aw = wr ? ((const float4*)wr)[cb / 4 + q4] : make_float4(0.f, 0.f, 0.f, 0.f);
        const int c = k * JSTR + cb + q4 * 4;
        sJ[w][c + 0] = f2bf(aj.x);
        sJ[w][c + 1] = f2bf(aj.y);
        sJ[w][c + 2] = f2bf(aj.z);
        sJ[w][c + 3] = f2bf(aj.w);
        sWt[w][c + 0] = f2bf(aw.x);
        sWt[w][c + 1] = f2bf(aw.y);
        sWt[w][c + 2] = f2bf(aw.z);
        sWt[w][c + 3] = f2bf(aw.w);
      }
    }
    {
      float ev0 = ev[(size_t)n * FDIM + lane];
      float ev1 = ev[(size_t)n * FDIM + lane + 32];
      float a0 = 0.f, a1 = 0.f;
#pragma unroll
      for (int f = 0; f < 64; ++f) {
        float evf = (f < 32) ? __shfl(ev0, f, 32) : __shfl(ev1, f - 32, 32);
        a0 += evf * bf2f(sW1a[f * 64 + lane]);
        a1 += evf * bf2f(sW1a[f * 64 + lane + 32]);
      }
      sAv[w][lane] = a0;
      sAv[w][lane + 32] = a1;
    }
    asm volatile("s_wait_dscnt 0" ::: "memory");

    v16bf AW0 = a_frag_lds(sWt[w], JSTR, 0, lane);
    v16bf AW1 = a_frag_lds(sWt[w], JSTR, 32, lane);
    v16bf AJ0 = a_frag_lds(sJ[w], JSTR, 0, lane);
    v16bf AJ1 = a_frag_lds(sJ[w], JSTR, 32, lane);

    float xp[8];
#pragma unroll
    for (int r = 0; r < 8; ++r) xp[r] = 0.f;

#pragma unroll
    for (int nt = 0; nt < 4; ++nt) {
      v8f C = {0.f, 0.f, 0.f, 0.f, 0.f, 0.f, 0.f, 0.f};
      C = wmma_bf16(AW0, BW1[0][nt], C);
      C = wmma_bf16(AW1, BW1[1][nt], C);
      C = wmma_bf16(AJ0, BW2[0][nt], C);
      C = wmma_bf16(AJ1, BW2[1][nt], C);
      const int col = nt * 16 + (lane & 15);
      const float add = sBias[col] + sAv[w][col];
      const float vv = sVvec[col];
#pragma unroll
      for (int r = 0; r < 8; ++r) xp[r] += reluf(C[r] + add) * vv;
    }
#pragma unroll
    for (int r = 0; r < 8; ++r) {
      float x = xp[r];
      x += __shfl_xor(x, 1, 32);
      x += __shfl_xor(x, 2, 32);
      x += __shfl_xor(x, 4, 32);
      x += __shfl_xor(x, 8, 32);
      xp[r] = x;
    }
    if ((lane & 15) == 0) {
      const int mb = (lane >> 4) * 8;
#pragma unroll
      for (int r = 0; r < 8; ++r) sX[w][mb + r] = xp[r];
    }
    asm volatile("s_wait_dscnt 0" ::: "memory");
    float xk = sX[w][lane & 15];
    float mx = xk;
    mx = fmaxf(mx, __shfl_xor(mx, 1, 32));
    mx = fmaxf(mx, __shfl_xor(mx, 2, 32));
    mx = fmaxf(mx, __shfl_xor(mx, 4, 32));
    mx = fmaxf(mx, __shfl_xor(mx, 8, 32));
    float e = __expf(xk - mx);
    float se = e;
    se += __shfl_xor(se, 1, 32);
    se += __shfl_xor(se, 2, 32);
    se += __shfl_xor(se, 4, 32);
    se += __shfl_xor(se, 8, 32);
    const float aww = e / se;
    float o0 = 0.f, o1 = 0.f;
#pragma unroll
    for (int k = 0; k < 16; ++k) {
      const float ak = __shfl(aww, k, 32);
      o0 += ak * bf2f(sJ[w][k * JSTR + lane]);
      o1 += ak * bf2f(sJ[w][k * JSTR + lane + 32]);
    }
    outp[(size_t)n * FDIM + lane] = o0;
    outp[(size_t)n * FDIM + lane + 32] = o1;
  }
}

// ---------------------------------------------------------------------------
// atten2 + conv, WMMA version: one wave per 16-node group.
//  - attention: per s, A_s[16x64] @ U (8 WMMA) -> relu(.)+q dot p -> softmax(3)
//  - vector convs: A2[16x192] @ Wv[192x48] (block-structured weights, 18 WMMA)
//  - bit conv (K=3): lane-parallel VALU
// Writes padded bf16 feature rows cbuf[N][576].
// ---------------------------------------------------------------------------
__global__ __launch_bounds__(128) void atten2_conv_kernel(
    const float* __restrict__ r0, const float* __restrict__ r1,
    const float* __restrict__ r2, const float* __restrict__ U,
    const float* __restrict__ qv, const float* __restrict__ pv,
    const float* __restrict__ Wbit, const float* __restrict__ Wv1,
    const float* __restrict__ Wv2, const float* __restrict__ Wv3,
    __bf16* __restrict__ cbuf) {
  __shared__ __bf16 sU[64 * 64];
  __shared__ __bf16 sWv[192 * 48];
  __shared__ float sWbit[24], sQ[64], sP[64];
  __shared__ __bf16 sR[4][3 * 16 * RSTR];  // [s][node][RSTR]
  __shared__ float sXs[4][64];             // xs[node*4+s]
  __shared__ float sBs[4][64];             // softmax factors

  const int tid = threadIdx.x;
  for (int i = tid; i < 64 * 64; i += 128) sU[i] = f2bf(U[i]);
  for (int i = tid; i < 192 * 48; i += 128) {
    const int k = i / 48, col = i % 48;
    const int s = k >> 6, f = k & 63;
    float val = 0.f;
    if (col < 24) {
      const int c = col / 3, sp = col % 3;
      if (s == sp) val = Wv1[c * 64 + f];
    } else if (col < 40) {
      const int j = col - 24, c = j >> 1, sp = j & 1;
      if (s == sp) val = Wv2[c * 128 + f];
      else if (s == sp + 1) val = Wv2[c * 128 + 64 + f];
    } else {
      const int c = col - 40;
      val = Wv3[c * 192 + s * 64 + f];
    }
    sWv[i] = f2bf(val);
  }
  for (int i = tid; i < 24; i += 128) sWbit[i] = Wbit[i];
  if (tid < 64) { sQ[tid] = qv[tid]; sP[tid] = pv[tid]; }
  __syncthreads();

  const int lane = tid & 31;
  const int w = tid >> 5;
  const int ngroups = (NNODES + 15) / 16;
  __bf16* sRw = sR[w];

  for (int g = blockIdx.x * 4 + w; g < ngroups; g += gridDim.x * 4) {
    // ---- stage 3 x 16 x 64 rows (zero-fill past N) ----
    const float* const srcs[3] = {r0, r1, r2};
#pragma unroll
    for (int t = 0; t < 3; ++t) {
      const int node = lane >> 1;
      const int halfc = (lane & 1) * 32;
      const int gn = g * 16 + node;
      const bool ok = gn < NNODES;
      const float* rp = srcs[t] + (size_t)gn * 64 + halfc;
#pragma unroll
      for (int q4 = 0; q4 < 8; ++q4) {
        float4 a4 = ok ? ((const float4*)rp)[q4] : make_float4(0.f, 0.f, 0.f, 0.f);
        const int base = (t * 16 + node) * RSTR + halfc + q4 * 4;
        sRw[base + 0] = f2bf(a4.x);
        sRw[base + 1] = f2bf(a4.y);
        sRw[base + 2] = f2bf(a4.z);
        sRw[base + 3] = f2bf(a4.w);
      }
    }
    asm volatile("s_wait_dscnt 0" ::: "memory");

    // ---- attention logits: per s, A_s @ U, relu(+q) dot p ----
    v16bf BU[2][4];
#pragma unroll
    for (int kc = 0; kc < 2; ++kc)
#pragma unroll
      for (int nt = 0; nt < 4; ++nt)
        BU[kc][nt] = b_frag_lds(sU, 64, kc * 32, nt * 16, lane);
#pragma unroll
    for (int s = 0; s < 3; ++s) {
      v16bf AF0 = a_frag_lds(sRw + (s * 16) * RSTR, RSTR, 0, lane);
      v16bf AF1 = a_frag_lds(sRw + (s * 16) * RSTR, RSTR, 32, lane);
      float xp[8];
#pragma unroll
      for (int r = 0; r < 8; ++r) xp[r] = 0.f;
#pragma unroll
      for (int nt = 0; nt < 4; ++nt) {
        v8f C = {0.f, 0.f, 0.f, 0.f, 0.f, 0.f, 0.f, 0.f};
        C = wmma_bf16(AF0, BU[0][nt], C);
        C = wmma_bf16(AF1, BU[1][nt], C);
        const int col = nt * 16 + (lane & 15);
        const float qq = sQ[col], pp = sP[col];
#pragma unroll
        for (int r = 0; r < 8; ++r) xp[r] += reluf(C[r] + qq) * pp;
      }
#pragma unroll
      for (int r = 0; r < 8; ++r) {
        float x = xp[r];
        x += __shfl_xor(x, 1, 32);
        x += __shfl_xor(x, 2, 32);
        x += __shfl_xor(x, 4, 32);
        x += __shfl_xor(x, 8, 32);
        xp[r] = x;
      }
      if ((lane & 15) == 0) {
        const int mb = (lane >> 4) * 8;
#pragma unroll
        for (int r = 0; r < 8; ++r) sXs[w][(mb + r) * 4 + s] = xp[r];
      }
    }
    asm volatile("s_wait_dscnt 0" ::: "memory");
    // ---- softmax over the 3 rows, per node ----
    if (lane < 16) {
      const float x0 = sXs[w][lane * 4 + 0];
      const float x1 = sXs[w][lane * 4 + 1];
      const float x2 = sXs[w][lane * 4 + 2];
      const float mx = fmaxf(x0, fmaxf(x1, x2));
      const float e0 = __expf(x0 - mx), e1 = __expf(x1 - mx), e2 = __expf(x2 - mx);
      const float inv = 1.f / (e0 + e1 + e2);
      sBs[w][lane * 4 + 0] = e0 * inv;
      sBs[w][lane * 4 + 1] = e1 * inv;
      sBs[w][lane * 4 + 2] = e2 * inv;
    }
    asm volatile("s_wait_dscnt 0" ::: "memory");
    // ---- scale rows in place: eN = b * uit ----
#pragma unroll
    for (int t = 0; t < 3; ++t) {
      const int node = lane >> 1;
      const int halfc = (lane & 1) * 32;
      const float bsv = sBs[w][node * 4 + t];
      const int base = (t * 16 + node) * RSTR + halfc;
#pragma unroll
      for (int i = 0; i < 32; ++i) sRw[base + i] = f2bf(bf2f(sRw[base + i]) * bsv);
    }
    asm volatile("s_wait_dscnt 0" ::: "memory");

    // ---- bit conv (K=3) + zero pad, lane-parallel VALU ----
    for (int node = 0; node < 16; ++node) {
      const int gn = g * 16 + node;
      if (gn < NNODES) {
        __bf16* co = cbuf + (size_t)gn * INKP;
        float rlo[3], rhi[3];
#pragma unroll
        for (int s = 0; s < 3; ++s) {
          rlo[s] = bf2f(sRw[(s * 16 + node) * RSTR + lane]);
          rhi[s] = bf2f(sRw[(s * 16 + node) * RSTR + lane + 32]);
        }
#pragma unroll
        for (int t = 0; t < 16; ++t) {
          const int c = t >> 1;
          const float* rr = (t & 1) ? rhi : rlo;
          const float vv = rr[0] * sWbit[c * 3] + rr[1] * sWbit[c * 3 + 1] +
                           rr[2] * sWbit[c * 3 + 2];
          co[c * 64 + ((t & 1) ? (lane + 32) : lane)] = f2bf(reluf(vv));
        }
        if (lane < 16) co[560 + lane] = f2bf(0.f);
      }
    }

    // ---- vector convs: A2[16x192] @ Wv[192x48] ----
    v16bf A2[6];
#pragma unroll
    for (int kc = 0; kc < 6; ++kc) {
      const int half = lane >> 4;
      const int m = lane & 15;
      v16bf a;
#pragma unroll
      for (int h = 0; h < 16; ++h) {
        const int gg = h >> 1, b = h & 1;
        int k = (gg < 4) ? (2 * gg + b) : (16 + 2 * (gg - 4) + b);
        k += half ? 8 : 0;
        const int kk = kc * 32 + k;
        a[h] = sRw[((kk >> 6) * 16 + m) * RSTR + (kk & 63)];
      }
      A2[kc] = a;
    }
#pragma unroll 1
    for (int nt = 0; nt < 3; ++nt) {
      v8f C = {0.f, 0.f, 0.f, 0.f, 0.f, 0.f, 0.f, 0.f};
#pragma unroll
      for (int kc = 0; kc < 6; ++kc) {
        v16bf bw = b_frag_lds(sWv, 48, kc * 32, nt * 16, lane);
        C = wmma_bf16(A2[kc], bw, C);
      }
      const int col = nt * 16 + (lane & 15);
#pragma unroll
      for (int r = 0; r < 8; ++r) {
        const int node = g * 16 + r + ((lane >> 4) ? 8 : 0);
        if (node < NNODES)
          cbuf[(size_t)node * INKP + 512 + col] = f2bf(reluf(C[r]));
      }
    }
  }
}

// ---------------------------------------------------------------------------
// final GEMM: relu(c[N,560] @ Wf[560,64] + bf) via bf16 WMMA.
// ---------------------------------------------------------------------------
__global__ __launch_bounds__(256) void final_gemm_kernel(
    const __bf16* __restrict__ cbuf, const float* __restrict__ Wf,
    const float* __restrict__ bfv, float* __restrict__ outp) {
  const int tid = threadIdx.x;
  const int lane = tid & 31;
  const int w = tid >> 5;
  const int ngroups = (NNODES + 15) / 16;
#pragma unroll 1
  for (int nt = 0; nt < 4; ++nt) {
    v16bf BW[18];
#pragma unroll
    for (int kc = 0; kc < 18; ++kc) {
      const int nn = nt * 16 + (lane & 15);
      const int kh = kc * 32 + ((lane >> 4) << 3);
      v16bf bfr;
#pragma unroll
      for (int h = 0; h < 16; ++h) {
        const int k = kh + h + (h >= 8 ? 8 : 0);
        bfr[h] = (k < INK) ? f2bf(Wf[(size_t)k * OUTF + nn]) : f2bf(0.f);
      }
      BW[kc] = bfr;
    }
    const float badd = bfv[nt * 16 + (lane & 15)];
    for (int g = blockIdx.x * 8 + w; g < ngroups; g += gridDim.x * 8) {
      const int m = lane & 15;
      const int node = g * 16 + m;
      const bool valid = node < NNODES;
      const __bf16* arow = cbuf + (size_t)node * INKP;
      const int half = lane >> 4;
      v8f C = {0.f, 0.f, 0.f, 0.f, 0.f, 0.f, 0.f, 0.f};
#pragma unroll 1
      for (int kc = 0; kc < 18; ++kc) {
        v16bf a;
#pragma unroll
        for (int h = 0; h < 16; ++h) {
          const int gg = h >> 1, b = h & 1;
          int k = (gg < 4) ? (2 * gg + b) : (16 + 2 * (gg - 4) + b);
          k += half ? 8 : 0;
          a[h] = valid ? arow[kc * 32 + k] : f2bf(0.f);
        }
        C = wmma_bf16(a, BW[kc], C);
      }
#pragma unroll
      for (int r = 0; r < 8; ++r) {
        const int mrow = r + (half ? 8 : 0);
        const int nod = g * 16 + mrow;
        if (nod < NNODES)
          outp[(size_t)nod * OUTF + nt * 16 + (lane & 15)] = reluf(C[r] + badd);
      }
    }
  }
}

extern "C" void kernel_launch(void* const* d_in, const int* in_sizes, int n_in,
                              void* d_out, int out_size, void* d_ws, size_t ws_size,
                              hipStream_t stream) {
  (void)in_sizes; (void)n_in; (void)out_size; (void)ws_size;
  const float* eu = (const float*)d_in[0];
  const float* ei = (const float*)d_in[1];
  const float* et = (const float*)d_in[2];
  const float* ew = (const float*)d_in[3];
  const int* u_iw_j = (const int*)d_in[4];
  const int* u_iw_w = (const int*)d_in[5];
  const int* u_tw_j = (const int*)d_in[6];
  const int* u_tw_w = (const int*)d_in[7];
  const int* i_uw_j = (const int*)d_in[8];
  const int* i_uw_w = (const int*)d_in[9];
  const int* i_tw_j = (const int*)d_in[10];
  const int* i_tw_w = (const int*)d_in[11];
  const int* t_uw_j = (const int*)d_in[12];
  const int* t_uw_w = (const int*)d_in[13];
  const int* t_iw_j = (const int*)d_in[14];
  const int* t_iw_w = (const int*)d_in[15];
  const float* W1_user = (const float*)d_in[16];
  const float* W2_user = (const float*)d_in[17];
  const float* b_user = (const float*)d_in[18];
  const float* v_user = (const float*)d_in[19];
  const float* W1_item = (const float*)d_in[20];
  const float* W2_item = (const float*)d_in[21];
  const float* b_item = (const float*)d_in[22];
  const float* v_item = (const float*)d_in[23];
  const float* W1_tag = (const float*)d_in[24];
  const float* W2_tag = (const float*)d_in[25];
  const float* b_tag = (const float*)d_in[26];
  const float* v_tag = (const float*)d_in[27];
  const float* U = (const float*)d_in[28];
  const float* q = (const float*)d_in[29];
  const float* p = (const float*)d_in[30];
  const float* W_bit = (const float*)d_in[31];
  const float* W_vec1 = (const float*)d_in[32];
  const float* W_vec2 = (const float*)d_in[33];
  const float* W_vec3 = (const float*)d_in[34];
  const float* Wf = (const float*)d_in[35];
  const float* bfp = (const float*)d_in[36];

  char* ws = (char*)d_ws;
  const size_t vecsz = (size_t)NNODES * FDIM;
  float* eu_iN = (float*)ws;
  float* eu_tN = eu_iN + vecsz;
  float* ei_uN = eu_tN + vecsz;
  float* ei_tN = ei_uN + vecsz;
  float* et_uN = ei_tN + vecsz;
  float* et_iN = et_uN + vecsz;
  __bf16* cu = (__bf16*)(et_iN + vecsz);
  __bf16* ci = cu + (size_t)NNODES * INKP;
  __bf16* ct = ci + (size_t)NNODES * INKP;

  const int g1 = 512;
  atten1_kernel<<<g1, 256, 0, stream>>>(eu, ei, ew, u_iw_j, u_iw_w, W1_item, W2_item, b_item, v_item, eu_iN);
  atten1_kernel<<<g1, 256, 0, stream>>>(eu, et, ew, u_tw_j, u_tw_w, W1_tag, W2_tag, b_tag, v_tag, eu_tN);
  atten1_kernel<<<g1, 256, 0, stream>>>(ei, eu, ew, i_uw_j, i_uw_w, W1_user, W2_user, b_user, v_user, ei_uN);
  atten1_kernel<<<g1, 256, 0, stream>>>(ei, et, ew, i_tw_j, i_tw_w, W1_tag, W2_tag, b_tag, v_tag, ei_tN);
  atten1_kernel<<<g1, 256, 0, stream>>>(et, eu, ew, t_uw_j, t_uw_w, W1_user, W2_user, b_user, v_user, et_uN);
  atten1_kernel<<<g1, 256, 0, stream>>>(et, ei, ew, t_iw_j, t_iw_w, W1_item, W2_item, b_item, v_item, et_iN);

  const int ngroups = (NNODES + 15) / 16;
  const int g2 = (ngroups + 3) / 4;
  atten2_conv_kernel<<<g2, 128, 0, stream>>>(eu, eu_iN, eu_tN, U, q, p, W_bit, W_vec1, W_vec2, W_vec3, cu);
  atten2_conv_kernel<<<g2, 128, 0, stream>>>(ei_uN, ei, ei_tN, U, q, p, W_bit, W_vec1, W_vec2, W_vec3, ci);
  atten2_conv_kernel<<<g2, 128, 0, stream>>>(et_uN, et_iN, et, U, q, p, W_bit, W_vec1, W_vec2, W_vec3, ct);

  float* out = (float*)d_out;
  const int g3 = 200;
  final_gemm_kernel<<<g3, 256, 0, stream>>>(cu, Wf, bfp, out);
  final_gemm_kernel<<<g3, 256, 0, stream>>>(ci, Wf, bfp, out + vecsz);
  final_gemm_kernel<<<g3, 256, 0, stream>>>(ct, Wf, bfp, out + 2 * vecsz);
}
